// LSTM_609885356153
// MI455X (gfx1250) — matrix operand order repaired and checked by reading, not compiled, not run on hardware
//
#include <hip/hip_runtime.h>
#include <hip/hip_bf16.h>

// ---------------------------------------------------------------------------
// Model dims (from reference): VOCAB=5000, EMBED=64, UNITS=64, B=256, T=2048
// ---------------------------------------------------------------------------
#define VOCAB_SZ 5000
#define EMBED_SZ 64
#define UNITS_SZ 64
#define B_SZ     256
#define T_SZ     2048
#define T2_SZ    128          // layer-2 sequence length = 2*UNITS
#define G4       256          // 4*UNITS gate width

typedef __attribute__((ext_vector_type(16))) _Float16 v16h;
typedef __attribute__((ext_vector_type(8)))  float    v8f;

__device__ __forceinline__ float sigm_(float x) {
    return 1.0f / (1.0f + __expf(-x));
}
__device__ __forceinline__ float tanh_(float x) {
    return 2.0f / (1.0f + __expf(-2.0f * x)) - 1.0f;
}

// ---------------------------------------------------------------------------
// A-fragment loader: 16x32 f16 A tile from LDS h-buffer (row-major [16][64]).
// ISA layout (CDNA5 ISA): lanes 0-15 -> M=lane, K(half-pairs) 0..7,16..23;
//                          lanes 16-31 -> M=lane-16, K 8..15,24..31.
// ---------------------------------------------------------------------------
__device__ __forceinline__ v16h load_a_frag(const _Float16* hb, int lane, int kbase) {
    v16h a;
    const int M    = lane & 15;
    const int koff = (lane < 16) ? 0 : 8;
#pragma unroll
    for (int p = 0; p < 8; ++p) {
        const int k = kbase + koff + ((p < 4) ? (2 * p) : (16 + 2 * (p - 4)));
        a[2 * p]     = hb[M * UNITS_SZ + k];
        a[2 * p + 1] = hb[M * UNITS_SZ + k + 1];
    }
    return a;
}

// ---------------------------------------------------------------------------
// B-fragment loader: 32x16 f16 B tile from global f32 U [64][256] row-major.
// Lanes 0-15 -> N=lane, K 0..7,16..23 (2 per VGPR); lanes 16-31 -> N=lane-16, K 8..15,24..31.
// Loaded ONCE and pinned in VGPRs for the whole time loop.
// ---------------------------------------------------------------------------
__device__ __forceinline__ v16h load_b_frag(const float* __restrict__ U,
                                            int lane, int kbase, int ncolbase) {
    v16h b;
    const int n    = ncolbase + (lane & 15);
    const int koff = kbase + ((lane < 16) ? 0 : 8);
#pragma unroll
    for (int p = 0; p < 8; ++p) {
        const int k = koff + ((p < 4) ? (2 * p) : (16 + 2 * (p - 4)));
        b[2 * p]     = (_Float16)U[(k)     * G4 + n];
        b[2 * p + 1] = (_Float16)U[(k + 1) * G4 + n];
    }
    return b;
}

// ---------------------------------------------------------------------------
// Kernel 0: gW[v][:] = embed[v]@W + b  (factorizes embedding through W; makes
// the per-timestep x@W term a 1KB L2 gather instead of a 34-GFLOP GEMM).
// grid (5000, 2), 256 threads.
// ---------------------------------------------------------------------------
__global__ __launch_bounds__(256) void embw_kernel(
    const float* __restrict__ embed,
    const float* __restrict__ W1f, const float* __restrict__ b1f,
    const float* __restrict__ W1b, const float* __restrict__ b1b,
    float* __restrict__ gWf, float* __restrict__ gWb)
{
    __shared__ float e[EMBED_SZ];
    const int v = blockIdx.x, dir = blockIdx.y, tid = threadIdx.x;
    const float* W = dir ? W1b : W1f;
    const float* b = dir ? b1b : b1f;
    float* gW = dir ? gWb : gWf;
    if (tid < EMBED_SZ) e[tid] = embed[v * EMBED_SZ + tid];
    __syncthreads();
    float s = b[tid];
#pragma unroll
    for (int k = 0; k < EMBED_SZ; ++k) s += e[k] * W[k * G4 + tid];
    gW[v * G4 + tid] = s;
}

// ---------------------------------------------------------------------------
// Kernel 1: layer-1 LSTM recurrence. 32 blocks = 16 batch tiles x 2 dirs.
// 256 threads = 8 wave32; wave w owns output column tiles 2w, 2w+1 with its
// U fragments pinned in VGPRs. Per step: 4 v_wmma_f32_16x16x32_f16 per wave.
// ---------------------------------------------------------------------------
__global__ __launch_bounds__(256) void lstm1_kernel(
    const int*   __restrict__ ids,
    const float* __restrict__ gWf, const float* __restrict__ gWb,
    const float* __restrict__ U1f, const float* __restrict__ U1b,
    float* __restrict__ h1)   // [B][128]: fwd cols 0..63, bwd cols 64..127
{
    __shared__ _Float16 hbuf[16 * UNITS_SZ];   // h state, f16, A-frag source
    __shared__ float    cbuf[16 * UNITS_SZ];   // c state
    __shared__ float    zbuf[16 * G4];         // recurrent pre-activations

    const int tid  = threadIdx.x;
    const int lane = tid & 31;
    const int wave = tid >> 5;
    const int dir  = blockIdx.x & 1;
    const int row0 = (blockIdx.x >> 1) * 16;

    const float* gW = dir ? gWb : gWf;
    const float* U  = dir ? U1b : U1f;

    for (int i = tid; i < 16 * UNITS_SZ; i += 256) {
        hbuf[i] = (_Float16)0.0f;
        cbuf[i] = 0.0f;
    }

    // Pin U fragments in VGPRs (constant over time).
    v16h bfr[2][2];
#pragma unroll
    for (int tile = 0; tile < 2; ++tile)
#pragma unroll
        for (int s = 0; s < 2; ++s)
            bfr[tile][s] = load_b_frag(U, lane, 32 * s, (wave * 2 + tile) * 16);

    const int lrow = tid >> 4;          // combine-phase local row 0..15
    const int cu0  = (tid & 15) * 4;    // combine-phase unit base
    const int crow = row0 + lrow;

    for (int t = 0; t < T_SZ; ++t) {
        __syncthreads();
        const int tt = dir ? (T_SZ - 1 - t) : t;

        // ---- recurrent GEMM: z = h @ U via WMMA ----
        const v16h a0 = load_a_frag(hbuf, lane, 0);
        const v16h a1 = load_a_frag(hbuf, lane, 32);
#pragma unroll
        for (int tile = 0; tile < 2; ++tile) {
            v8f acc = {};
            acc = __builtin_amdgcn_wmma_f32_16x16x32_f16(
                false, a0, false, bfr[tile][0], (short)0, acc, false, false);
            acc = __builtin_amdgcn_wmma_f32_16x16x32_f16(
                false, a1, false, bfr[tile][1], (short)0, acc, false, false);
            const int nb   = (wave * 2 + tile) * 16 + (lane & 15);
            const int mofs = (lane < 16) ? 0 : 8;   // C layout: VGPR r -> M=r(+8)
#pragma unroll
            for (int r = 0; r < 8; ++r)
                zbuf[(r + mofs) * G4 + nb] = acc[r];
        }
        __syncthreads();

        // ---- gate combine: fuse gather of (embed@W+b)[id], mask carry ----
        const int   idv   = ids[crow * T_SZ + tt];
        const float* gWr  = gW + (long)idv * G4;
        // prefetch next timestep's gather row (global_prefetch_b8)
        const int tn = tt + (dir ? -1 : 1);
        if (((tid & 15) == 0) && ((unsigned)tn < (unsigned)T_SZ)) {
            const int idn = ids[crow * T_SZ + tn];
            __builtin_prefetch(gW + (long)idn * G4, 0, 0);
        }
        const bool m = (idv != 0);
#pragma unroll
        for (int j = 0; j < 4; ++j) {
            const int u  = cu0 + j;
            const float zi = zbuf[lrow * G4 + u      ] + gWr[u      ];
            const float zf = zbuf[lrow * G4 + u +  64] + gWr[u +  64];
            const float zg = zbuf[lrow * G4 + u + 128] + gWr[u + 128];
            const float zo = zbuf[lrow * G4 + u + 192] + gWr[u + 192];
            const float c  = cbuf[lrow * UNITS_SZ + u];
            const float cn = sigm_(zf) * c + sigm_(zi) * tanh_(zg);
            const float hn = sigm_(zo) * tanh_(cn);
            if (m) {
                cbuf[lrow * UNITS_SZ + u] = cn;
                hbuf[lrow * UNITS_SZ + u] = (_Float16)hn;
            }
        }
    }
    __syncthreads();
#pragma unroll
    for (int j = 0; j < 4; ++j) {
        const int u = cu0 + j;
        h1[(crow) * (2 * UNITS_SZ) + dir * UNITS_SZ + u] =
            (float)hbuf[lrow * UNITS_SZ + u];
    }
}

// ---------------------------------------------------------------------------
// Kernel 2: layer-2 LSTM over the 128 features of h1 (input dim = 1).
// Same WMMA recurrence; input term x*W2+b2 fused into combine. mask all-true.
// ---------------------------------------------------------------------------
__global__ __launch_bounds__(256) void lstm2_kernel(
    const float* __restrict__ h1,    // [B][128] (the layer-2 "sequence")
    const float* __restrict__ W2f, const float* __restrict__ b2f,
    const float* __restrict__ W2b, const float* __restrict__ b2b,
    const float* __restrict__ U2f, const float* __restrict__ U2b,
    float* __restrict__ h2)          // [B][128]
{
    __shared__ _Float16 hbuf[16 * UNITS_SZ];
    __shared__ float    cbuf[16 * UNITS_SZ];
    __shared__ float    zbuf[16 * G4];

    const int tid  = threadIdx.x;
    const int lane = tid & 31;
    const int wave = tid >> 5;
    const int dir  = blockIdx.x & 1;
    const int row0 = (blockIdx.x >> 1) * 16;

    const float* W2 = dir ? W2b : W2f;
    const float* b2 = dir ? b2b : b2f;
    const float* U  = dir ? U2b : U2f;

    for (int i = tid; i < 16 * UNITS_SZ; i += 256) {
        hbuf[i] = (_Float16)0.0f;
        cbuf[i] = 0.0f;
    }

    v16h bfr[2][2];
#pragma unroll
    for (int tile = 0; tile < 2; ++tile)
#pragma unroll
        for (int s = 0; s < 2; ++s)
            bfr[tile][s] = load_b_frag(U, lane, 32 * s, (wave * 2 + tile) * 16);

    const int lrow = tid >> 4;
    const int cu0  = (tid & 15) * 4;
    const int crow = row0 + lrow;

    for (int t = 0; t < T2_SZ; ++t) {
        __syncthreads();
        const int s = dir ? (T2_SZ - 1 - t) : t;

        const v16h a0 = load_a_frag(hbuf, lane, 0);
        const v16h a1 = load_a_frag(hbuf, lane, 32);
#pragma unroll
        for (int tile = 0; tile < 2; ++tile) {
            v8f acc = {};
            acc = __builtin_amdgcn_wmma_f32_16x16x32_f16(
                false, a0, false, bfr[tile][0], (short)0, acc, false, false);
            acc = __builtin_amdgcn_wmma_f32_16x16x32_f16(
                false, a1, false, bfr[tile][1], (short)0, acc, false, false);
            const int nb   = (wave * 2 + tile) * 16 + (lane & 15);
            const int mofs = (lane < 16) ? 0 : 8;
#pragma unroll
            for (int r = 0; r < 8; ++r)
                zbuf[(r + mofs) * G4 + nb] = acc[r];
        }
        __syncthreads();

        const float x = h1[crow * T2_SZ + s];   // scalar input this step
#pragma unroll
        for (int j = 0; j < 4; ++j) {
            const int u  = cu0 + j;
            const float zi = zbuf[lrow * G4 + u      ] + x * W2[u      ] + b2[u      ];
            const float zf = zbuf[lrow * G4 + u +  64] + x * W2[u +  64] + b2[u +  64];
            const float zg = zbuf[lrow * G4 + u + 128] + x * W2[u + 128] + b2[u + 128];
            const float zo = zbuf[lrow * G4 + u + 192] + x * W2[u + 192] + b2[u + 192];
            const float c  = cbuf[lrow * UNITS_SZ + u];
            const float cn = sigm_(zf) * c + sigm_(zi) * tanh_(zg);
            const float hn = sigm_(zo) * tanh_(cn);
            cbuf[lrow * UNITS_SZ + u] = cn;
            hbuf[lrow * UNITS_SZ + u] = (_Float16)hn;
        }
    }
    __syncthreads();
#pragma unroll
    for (int j = 0; j < 4; ++j) {
        const int u = cu0 + j;
        h2[crow * (2 * UNITS_SZ) + dir * UNITS_SZ + u] =
            (float)hbuf[lrow * UNITS_SZ + u];
    }
}

// ---------------------------------------------------------------------------
// Kernel 3: dense head. 1 block, 256 threads (one batch row each).
// relu(h2@Wd1+bd1) -> relu(@Wd2+bd2) -> sigmoid(@Wo+bo)
// ---------------------------------------------------------------------------
__global__ __launch_bounds__(256) void dense_kernel(
    const float* __restrict__ h2,
    const float* __restrict__ Wd1, const float* __restrict__ bd1,
    const float* __restrict__ Wd2, const float* __restrict__ bd2,
    const float* __restrict__ Wo,  const float* __restrict__ bo,
    float* __restrict__ out)
{
    __shared__ float sW1[128 * 64];
    __shared__ float sW2[64 * 64];
    __shared__ float sb1[64], sb2[64], sWo[64];
    const int tid = threadIdx.x;
    for (int i = tid; i < 128 * 64; i += 256) sW1[i] = Wd1[i];
    for (int i = tid; i < 64 * 64;  i += 256) sW2[i] = Wd2[i];
    if (tid < 64) { sb1[tid] = bd1[tid]; sb2[tid] = bd2[tid]; sWo[tid] = Wo[tid]; }
    __syncthreads();

    const float* xr = h2 + tid * 128;   // this row's h2 (L1/L2 resident)
    float d1[64];
#pragma unroll 4
    for (int j = 0; j < 64; ++j) {
        float s = sb1[j];
        for (int k = 0; k < 128; ++k) s += xr[k] * sW1[k * 64 + j];
        d1[j] = fmaxf(s, 0.0f);
    }
    float o = bo[0];
#pragma unroll 4
    for (int j = 0; j < 64; ++j) {
        float s = sb2[j];
        for (int k = 0; k < 64; ++k) s += d1[k] * sW2[k * 64 + j];
        o += fmaxf(s, 0.0f) * sWo[j];
    }
    out[tid] = sigm_(o);
}

// ---------------------------------------------------------------------------
extern "C" void kernel_launch(void* const* d_in, const int* in_sizes, int n_in,
                              void* d_out, int out_size, void* d_ws, size_t ws_size,
                              hipStream_t stream)
{
    (void)in_sizes; (void)n_in; (void)out_size; (void)ws_size;

    const int*   ids   = (const int*)  d_in[0];
    const float* embed = (const float*)d_in[1];
    const float* W1f   = (const float*)d_in[2];
    const float* U1f   = (const float*)d_in[3];
    const float* b1f   = (const float*)d_in[4];
    const float* W1b   = (const float*)d_in[5];
    const float* U1b   = (const float*)d_in[6];
    const float* b1b   = (const float*)d_in[7];
    const float* W2f   = (const float*)d_in[8];
    const float* U2f   = (const float*)d_in[9];
    const float* b2f   = (const float*)d_in[10];
    const float* W2b   = (const float*)d_in[11];
    const float* U2b   = (const float*)d_in[12];
    const float* b2b   = (const float*)d_in[13];
    const float* Wd1   = (const float*)d_in[14];
    const float* bd1   = (const float*)d_in[15];
    const float* Wd2   = (const float*)d_in[16];
    const float* bd2   = (const float*)d_in[17];
    const float* Wo    = (const float*)d_in[18];
    const float* bo    = (const float*)d_in[19];

    float* ws  = (float*)d_ws;
    float* gWf = ws;                               // [5000][256]
    float* gWb = gWf + (size_t)VOCAB_SZ * G4;      // [5000][256]
    float* h1  = gWb + (size_t)VOCAB_SZ * G4;      // [256][128]
    float* h2  = h1  + (size_t)B_SZ * 2 * UNITS_SZ;// [256][128]

    embw_kernel<<<dim3(VOCAB_SZ, 2), 256, 0, stream>>>(
        embed, W1f, b1f, W1b, b1b, gWf, gWb);

    lstm1_kernel<<<32, 256, 0, stream>>>(ids, gWf, gWb, U1f, U1b, h1);

    lstm2_kernel<<<32, 256, 0, stream>>>(h1, W2f, b2f, W2b, b2b, U2f, U2b, h2);

    dense_kernel<<<1, 256, 0, stream>>>(h2, Wd1, bd1, Wd2, bd2, Wo, bo,
                                        (float*)d_out);
}
